// AttnBlockWithText_7619271983719
// MI455X (gfx1250) — compile-verified, matrix operand order checked by reading, not run on hardware
//
#include <hip/hip_runtime.h>
#include <hip/hip_bf16.h>

typedef _Float16 h16;
typedef __attribute__((ext_vector_type(16))) _Float16 v16h;
typedef __attribute__((ext_vector_type(8)))  float    v8f;
typedef __attribute__((ext_vector_type(4)))  unsigned int u32x4;

union FragV { u32x4 q[2]; v16h h; };

// Async copy global -> LDS (CDNA5 GLOBAL_LOAD_ASYNC_TO_LDS_B128, tracked by ASYNCcnt).
// ldsOff = byte offset within wave's LDS allocation (generic shared addr truncated to 32b).
__device__ __forceinline__ void async_copy_b128(unsigned ldsOff, const void* gaddr) {
  asm volatile("global_load_async_to_lds_b128 %0, %1, off"
               :: "v"(ldsOff), "v"(gaddr) : "memory");
}
__device__ __forceinline__ void wait_async0() {
  asm volatile("s_wait_asynccnt 0" ::: "memory");
}

// -------------------- GroupNorm statistics --------------------
// grid: 256 blocks = (b,g), 256 threads. Group g covers channels [24g,24g+24).
__global__ __launch_bounds__(256) void gn_stats_kernel(
    const float* __restrict__ x, const float* __restrict__ text,
    float* __restrict__ stats) {
  const int b = blockIdx.x >> 5;
  const int g = blockIdx.x & 31;
  const int tid = threadIdx.x;
  float s = 0.f, ss = 0.f;
  for (int idx = tid; idx < 24 * 1024; idx += 256) {
    const int c = g * 24 + (idx >> 10);
    const int p = idx & 1023;
    float v = (c < 256) ? x[(((size_t)b * 256 + c) << 10) + p]
                        : text[b * 512 + (c - 256)];
    s += v; ss += v * v;
  }
  for (int off = 16; off >= 1; off >>= 1) {
    s  += __shfl_xor(s,  off, 32);
    ss += __shfl_xor(ss, off, 32);
  }
  __shared__ float rs[8], rss[8];
  if ((tid & 31) == 0) { rs[tid >> 5] = s; rss[tid >> 5] = ss; }
  __syncthreads();
  if (tid == 0) {
    float S = 0.f, SS = 0.f;
    for (int i = 0; i < 8; ++i) { S += rs[i]; SS += rss[i]; }
    const float mean = S * (1.f / 24576.f);
    const float var  = SS * (1.f / 24576.f) - mean * mean;
    stats[blockIdx.x]       = mean;
    stats[256 + blockIdx.x] = rsqrtf(var + 1e-6f);
  }
}

// -------------------- Normalize + cast to f16, layout [B*HW, 768] --------------------
__global__ __launch_bounds__(256) void normalize_kernel(
    const float* __restrict__ x, const float* __restrict__ text,
    const float* __restrict__ stats, const float* __restrict__ gamma,
    const float* __restrict__ beta, h16* __restrict__ hn) {
  const size_t i = (size_t)blockIdx.x * 256 + threadIdx.x;  // 8*1024*768, c fastest
  const int c = (int)(i % 768);
  const size_t t = i / 768;
  const int p = (int)(t & 1023);
  const int b = (int)(t >> 10);
  const int g = c / 24;
  float v = (c < 256) ? x[(((size_t)b * 256 + c) << 10) + p]
                      : text[b * 512 + (c - 256)];
  const float mean = stats[b * 32 + g];
  const float rstd = stats[256 + b * 32 + g];
  hn[i] = (h16)(((v - mean) * rstd) * gamma[c] + beta[c]);
}

// -------------------- Weights -> f16, transposed [N=768][K=768]; bias concat --------------------
__global__ __launch_bounds__(256) void wconv_kernel(
    const float* __restrict__ W0, const float* __restrict__ W1, const float* __restrict__ W2,
    const float* __restrict__ b0, const float* __restrict__ b1, const float* __restrict__ b2,
    h16* __restrict__ Wt, float* __restrict__ bcat) {
  const int i = blockIdx.x * 256 + threadIdx.x;  // 768*768
  const int k = i % 768;
  const int n = i / 768;
  const float* W = (n < 256) ? W0 : ((n < 512) ? W1 : W2);
  const int nn = n & 255;
  Wt[(size_t)n * 768 + k] = (h16)W[(size_t)k * 256 + nn];
  if (k == 0)
    bcat[n] = (n < 256) ? b0[nn] : ((n < 512) ? b1[nn] : b2[nn]);
}

// -------------------- Fused QKV GEMM: [8192,768] x [768,768] -> qkv f16 --------------------
// Block tile 128x128; 8 waves 4(M)x2(N); wave tile 32x64 = 2x4 wmma tiles.
// Double-buffered GLOBAL_LOAD_ASYNC_TO_LDS_B128 tile staging.
__global__ __launch_bounds__(256) void gemm_qkv_kernel(
    const h16* __restrict__ A, const h16* __restrict__ Bt,
    const float* __restrict__ bias, h16* __restrict__ Cout) {
  __shared__ __align__(16) h16 As[2][128 * 48];  // stride 48 h16 = 96B, 16B aligned
  __shared__ __align__(16) h16 Bs[2][128 * 48];
  const int mBase = blockIdx.x * 128;
  const int nBase = blockIdx.y * 128;
  const int tid  = threadIdx.x;
  const int wave = tid >> 5;
  const int lane = tid & 31;
  const int wm = wave >> 1;       // 0..3
  const int wn = wave & 1;        // 0..1
  const int l16 = lane & 15;
  const int hi  = lane >> 4;      // 0/1 selects K phase per WMMA A/B layout

  const int row0 = tid >> 2;            // chunk id tid   -> row, seg
  const int seg0 = tid & 3;
  const int row1 = (tid + 256) >> 2;    // chunk id tid+256
  const int seg1 = (tid + 256) & 3;

  v8f acc[2][4];
  for (int mt = 0; mt < 2; ++mt)
    for (int nt = 0; nt < 4; ++nt)
      for (int r = 0; r < 8; ++r) acc[mt][nt][r] = 0.f;

  auto load_tiles = [&](int k0, int buf) {
    async_copy_b128((unsigned)(uintptr_t)&As[buf][row0 * 48 + seg0 * 8],
                    A + (size_t)(mBase + row0) * 768 + k0 + seg0 * 8);
    async_copy_b128((unsigned)(uintptr_t)&As[buf][row1 * 48 + seg1 * 8],
                    A + (size_t)(mBase + row1) * 768 + k0 + seg1 * 8);
    async_copy_b128((unsigned)(uintptr_t)&Bs[buf][row0 * 48 + seg0 * 8],
                    Bt + (size_t)(nBase + row0) * 768 + k0 + seg0 * 8);
    async_copy_b128((unsigned)(uintptr_t)&Bs[buf][row1 * 48 + seg1 * 8],
                    Bt + (size_t)(nBase + row1) * 768 + k0 + seg1 * 8);
  };

  load_tiles(0, 0);
  wait_async0();
  __syncthreads();

  int buf = 0;
  for (int k0 = 0; k0 < 768; k0 += 32) {
    if (k0 + 32 < 768) load_tiles(k0 + 32, buf ^ 1);  // overlap with compute

    FragV afr[2];
    for (int mt = 0; mt < 2; ++mt) {
      const int row = wm * 32 + mt * 16 + l16;
      const int ko  = hi * 8;
      afr[mt].q[0] = *(const u32x4*)(&As[buf][row * 48 + ko]);
      afr[mt].q[1] = *(const u32x4*)(&As[buf][row * 48 + ko + 16]);
    }
    for (int nt = 0; nt < 4; ++nt) {
      const int col = wn * 64 + nt * 16 + l16;
      const int ko  = hi * 8;
      FragV bfr;
      bfr.q[0] = *(const u32x4*)(&Bs[buf][col * 48 + ko]);
      bfr.q[1] = *(const u32x4*)(&Bs[buf][col * 48 + ko + 16]);
      for (int mt = 0; mt < 2; ++mt)
        acc[mt][nt] = __builtin_amdgcn_wmma_f32_16x16x32_f16(
            false, afr[mt].h, false, bfr.h, (short)0, acc[mt][nt], false, false);
    }

    wait_async0();
    __syncthreads();
    buf ^= 1;
  }

  for (int mt = 0; mt < 2; ++mt)
    for (int nt = 0; nt < 4; ++nt) {
      const int n = nBase + wn * 64 + nt * 16 + l16;
      const float bv = bias[n];
      for (int r = 0; r < 8; ++r) {
        const int m = mBase + wm * 32 + mt * 16 + r + hi * 8;
        Cout[(size_t)m * 768 + n] = (h16)(acc[mt][nt][r] + bv);
      }
    }
}

// -------------------- V transpose: vT[bh][ch=64][pos=1024] --------------------
__global__ __launch_bounds__(256) void vtrans_kernel(
    const h16* __restrict__ qkv, h16* __restrict__ vT) {
  const int i = blockIdx.x * 256 + threadIdx.x;  // 32*64*1024, pos fastest
  const int p  = i & 1023;
  const int t  = i >> 10;
  const int ch = t & 63;
  const int bh = t >> 6;
  const int b = bh >> 2, h = bh & 3;
  vT[i] = qkv[((size_t)b * 1024 + p) * 768 + 512 + h * 64 + ch];
}

// -------------------- Flash attention + residual --------------------
// grid: (32 bh, 4 query blocks). 8 waves/block, each wave owns 32 query rows.
__global__ __launch_bounds__(256) void attn_kernel(
    const h16* __restrict__ qkv, const h16* __restrict__ vT,
    const float* __restrict__ x, float* __restrict__ out) {
  __shared__ __align__(16) h16 Pls[8][32 * 72];  // per-wave P tile, stride 72 h16 = 144B
  const int bh = blockIdx.x;
  const int b = bh >> 2, h = bh & 3;
  const int tid = threadIdx.x, wave = tid >> 5, lane = tid & 31;
  const int l16 = lane & 15;
  const int hi  = lane >> 4;
  const int hi8 = hi * 8;
  const int qrow0 = blockIdx.y * 256 + wave * 32;
  h16* P = &Pls[wave][0];

  // Preload Q fragments: [mt][kc] covers 32 rows x 64 channels
  FragV qf[2][2];
  for (int mt = 0; mt < 2; ++mt)
    for (int kc = 0; kc < 2; ++kc) {
      const int row = qrow0 + mt * 16 + l16;
      const h16* base = qkv + ((size_t)b * 1024 + row) * 768 + h * 64 + kc * 32 + hi * 8;
      qf[mt][kc].q[0] = *(const u32x4*)(base);
      qf[mt][kc].q[1] = *(const u32x4*)(base + 16);
    }

  float mrow[2][8], lrow[2][8];
  v8f oacc[2][4];
  for (int mt = 0; mt < 2; ++mt)
    for (int r = 0; r < 8; ++r) { mrow[mt][r] = -1e30f; lrow[mt][r] = 0.f; }
  for (int mt = 0; mt < 2; ++mt)
    for (int nt = 0; nt < 4; ++nt)
      for (int r = 0; r < 8; ++r) oacc[mt][nt][r] = 0.f;

  for (int kt = 0; kt < 16; ++kt) {
    const int kbase = kt * 64;

    // Prefetch next key tile's K rows and V rows into cache (global_prefetch_b8)
    if (kt + 1 < 16) {
      const int nb = kbase + 64;
      __builtin_prefetch(qkv + ((size_t)b * 1024 + nb + lane) * 768 + 256 + h * 64, 0, 0);
      __builtin_prefetch(qkv + ((size_t)b * 1024 + nb + 32 + lane) * 768 + 256 + h * 64, 0, 0);
      __builtin_prefetch(vT + ((size_t)bh * 64 + lane) * 1024 + nb, 0, 0);
      __builtin_prefetch(vT + ((size_t)bh * 64 + 32 + lane) * 1024 + nb, 0, 0);
    }

    // ---- S = Q * K^T for a 32x64 score tile ----
    v8f sacc[2][4];
    for (int mt = 0; mt < 2; ++mt)
      for (int nt = 0; nt < 4; ++nt)
        for (int r = 0; r < 8; ++r) sacc[mt][nt][r] = 0.f;

    for (int nt = 0; nt < 4; ++nt) {
      const int kp = kbase + nt * 16 + l16;
      const h16* kb = qkv + ((size_t)b * 1024 + kp) * 768 + 256 + h * 64 + hi * 8;
      FragV kf0, kf1;
      kf0.q[0] = *(const u32x4*)(kb);       kf0.q[1] = *(const u32x4*)(kb + 16);
      kf1.q[0] = *(const u32x4*)(kb + 32);  kf1.q[1] = *(const u32x4*)(kb + 48);
      for (int mt = 0; mt < 2; ++mt) {
        sacc[mt][nt] = __builtin_amdgcn_wmma_f32_16x16x32_f16(
            false, qf[mt][0].h, false, kf0.h, (short)0, sacc[mt][nt], false, false);
        sacc[mt][nt] = __builtin_amdgcn_wmma_f32_16x16x32_f16(
            false, qf[mt][1].h, false, kf1.h, (short)0, sacc[mt][nt], false, false);
      }
    }

    // ---- scale + online softmax (row ops across 16-lane groups) ----
    for (int mt = 0; mt < 2; ++mt) {
      for (int nt = 0; nt < 4; ++nt)
        for (int r = 0; r < 8; ++r) sacc[mt][nt][r] *= 0.125f;  // Ch^-0.5

      for (int r = 0; r < 8; ++r) {
        float v = sacc[mt][0][r];
        for (int nt = 1; nt < 4; ++nt) v = fmaxf(v, sacc[mt][nt][r]);
        for (int off = 8; off >= 1; off >>= 1) v = fmaxf(v, __shfl_xor(v, off, 16));
        const float mold = mrow[mt][r];
        const float mnew = fmaxf(mold, v);
        const float corr = __expf(mold - mnew);
        mrow[mt][r] = mnew;
        float rsum = 0.f;
        for (int nt = 0; nt < 4; ++nt) {
          const float p = __expf(sacc[mt][nt][r] - mnew);
          sacc[mt][nt][r] = p;
          rsum += p;
        }
        for (int off = 8; off >= 1; off >>= 1) rsum += __shfl_xor(rsum, off, 16);
        lrow[mt][r] = lrow[mt][r] * corr + rsum;
        for (int nt = 0; nt < 4; ++nt) oacc[mt][nt][r] *= corr;
      }
      // C-layout -> LDS (row-major P tile) for A-layout reload
      for (int nt = 0; nt < 4; ++nt)
        for (int r = 0; r < 8; ++r)
          P[(mt * 16 + r + hi8) * 72 + nt * 16 + l16] = (h16)sacc[mt][nt][r];
    }
    __syncthreads();

    // ---- O += P * V ----
    for (int kc = 0; kc < 2; ++kc) {
      FragV pf[2];
      for (int mt = 0; mt < 2; ++mt) {
        const h16* pb = P + (mt * 16 + l16) * 72 + kc * 32 + hi * 8;
        pf[mt].q[0] = *(const u32x4*)(pb);
        pf[mt].q[1] = *(const u32x4*)(pb + 16);
      }
      for (int nt = 0; nt < 4; ++nt) {
        const int ch = nt * 16 + l16;
        const h16* vb = vT + ((size_t)bh * 64 + ch) * 1024 + kbase + kc * 32 + hi * 8;
        FragV vf;
        vf.q[0] = *(const u32x4*)(vb);
        vf.q[1] = *(const u32x4*)(vb + 16);
        for (int mt = 0; mt < 2; ++mt)
          oacc[mt][nt] = __builtin_amdgcn_wmma_f32_16x16x32_f16(
              false, pf[mt].h, false, vf.h, (short)0, oacc[mt][nt], false, false);
      }
    }
    __syncthreads();
  }

  // ---- finalize: O * (1/l), residual add, store f32 [B,C,H*W] ----
  float linv[2][8];
  for (int mt = 0; mt < 2; ++mt)
    for (int r = 0; r < 8; ++r) linv[mt][r] = 1.0f / lrow[mt][r];
  for (int mt = 0; mt < 2; ++mt)
    for (int nt = 0; nt < 4; ++nt) {
      const int c = h * 64 + nt * 16 + l16;
      for (int r = 0; r < 8; ++r) {
        const int pos = qrow0 + mt * 16 + r + hi8;
        const size_t idx = ((size_t)b * 256 + c) * 1024 + pos;
        out[idx] = x[idx] + oacc[mt][nt][r] * linv[mt][r];
      }
    }
}

// -------------------- launch --------------------
extern "C" void kernel_launch(void* const* d_in, const int* in_sizes, int n_in,
                              void* d_out, int out_size, void* d_ws, size_t ws_size,
                              hipStream_t stream) {
  const float* x     = (const float*)d_in[0];
  const float* text  = (const float*)d_in[1];
  const float* gamma = (const float*)d_in[2];
  const float* beta  = (const float*)d_in[3];
  const float* W0    = (const float*)d_in[4];
  const float* b0    = (const float*)d_in[5];
  const float* W1    = (const float*)d_in[6];
  const float* b1    = (const float*)d_in[7];
  const float* W2    = (const float*)d_in[8];
  const float* b2    = (const float*)d_in[9];
  float* out = (float*)d_out;

  char* ws = (char*)d_ws;
  float* stats = (float*)ws;                    // 512 f32
  float* bcat  = (float*)(ws + 2048);           // 768 f32
  h16*   hn    = (h16*)(ws + 8192);             // 8*1024*768
  h16*   Wt    = hn + (size_t)8 * 1024 * 768;   // 768*768
  h16*   qkv   = Wt + (size_t)768 * 768;        // 8*1024*768
  h16*   vT    = qkv + (size_t)8 * 1024 * 768;  // 32*64*1024

  gn_stats_kernel<<<256, 256, 0, stream>>>(x, text, stats);
  normalize_kernel<<<(8 * 1024 * 768) / 256, 256, 0, stream>>>(x, text, stats, gamma, beta, hn);
  wconv_kernel<<<(768 * 768) / 256, 256, 0, stream>>>(W0, W1, W2, b0, b1, b2, Wt, bcat);
  gemm_qkv_kernel<<<dim3(64, 6), 256, 0, stream>>>(hn, Wt, bcat, qkv);
  vtrans_kernel<<<(32 * 64 * 1024) / 256, 256, 0, stream>>>(qkv, vT);
  attn_kernel<<<dim3(32, 4), 256, 0, stream>>>(qkv, vT, x, out);
}
